// PointerGeneratorNetwork_4389456576578
// MI455X (gfx1250) — matrix operand order repaired and checked by reading, compiled, not run on hardware
//
#include <hip/hip_runtime.h>
#include <math.h>

// Problem sizes (match reference)
constexpr int B_ = 8;
constexpr int S_ = 1024;
constexpr int L_ = 2048;
constexpr int D_ = 1024;

typedef float  v8f    __attribute__((ext_vector_type(8)));
typedef __bf16 v16bf  __attribute__((ext_vector_type(16)));
typedef __bf16 bf16x8 __attribute__((ext_vector_type(8)));
typedef __bf16 bf16x4 __attribute__((ext_vector_type(4)));

// Native f32->bf16 (fptrunc) -- backend emits v_cvt_pk_bf16_f32 pairs.
__device__ __forceinline__ __bf16 f2bf(float f) { return (__bf16)f; }

__device__ __forceinline__ float waveReduceAdd(float v) {
#pragma unroll
    for (int off = 16; off > 0; off >>= 1) v += __shfl_down(v, off, 32);
    return v;
}

// ---------------------------------------------------------------------------
// K0: dec_pgen[b,s] = dec[b,s,:] . (w_pgen[:D] + w_pgen[2D:])
// ---------------------------------------------------------------------------
__global__ __launch_bounds__(128)
void pgn_dec_pgen_kernel(const float* __restrict__ dec,
                         const float* __restrict__ wpg,
                         float* __restrict__ dpg) {
    const int row = blockIdx.x;            // b*S + s
    const int tid = threadIdx.x;
    const float* r = dec + (size_t)row * D_;
    float acc = 0.f;
#pragma unroll
    for (int j = 0; j < D_ / 128; ++j) {
        int idx = j * 128 + tid;
        acc += r[idx] * (wpg[idx] + wpg[2 * D_ + idx]);
    }
    acc = waveReduceAdd(acc);
    __shared__ float red[4];
    if ((tid & 31) == 0) red[tid >> 5] = acc;
    __syncthreads();
    if (tid == 0) dpg[row] = red[0] + red[1] + red[2] + red[3];
}

// ---------------------------------------------------------------------------
// K1: es[b,l] = mask ? enc[b,l,:].w_ptr[D:] + b_ptr : -1e9
// ---------------------------------------------------------------------------
__global__ __launch_bounds__(128)
void pgn_enc_score_kernel(const float* __restrict__ enc,
                          const float* __restrict__ wptr,
                          const int* __restrict__ mask,
                          const float* __restrict__ bptr,
                          float* __restrict__ es) {
    const int row = blockIdx.x;            // b*L + l
    const int tid = threadIdx.x;
    const float* r = enc + (size_t)row * D_;
    const float* w = wptr + D_;
    float acc = 0.f;
#pragma unroll
    for (int j = 0; j < D_ / 128; ++j) {
        int idx = j * 128 + tid;
        acc += r[idx] * w[idx];
    }
    acc = waveReduceAdd(acc);
    __shared__ float red[4];
    if ((tid & 31) == 0) red[tid >> 5] = acc;
    __syncthreads();
    if (tid == 0) {
        float v = red[0] + red[1] + red[2] + red[3] + bptr[0];
        es[row] = (mask[row] != 0) ? v : -1e9f;
    }
}

// ---------------------------------------------------------------------------
// K2: p[b,:] = softmax(es[b,:]) over L   (softmax is shift-invariant, so the
// per-(b,s) dec_score term cancels exactly; one row per batch suffices)
// ---------------------------------------------------------------------------
__global__ __launch_bounds__(256)
void pgn_softmax_kernel(const float* __restrict__ es, float* __restrict__ p) {
    const int b = blockIdx.x;
    const int tid = threadIdx.x;
    const float* row = es + (size_t)b * L_;
    __shared__ float redm[8];
    __shared__ float reds[8];

    float m = -INFINITY;
    for (int i = tid; i < L_; i += 256) m = fmaxf(m, row[i]);
#pragma unroll
    for (int off = 16; off > 0; off >>= 1) m = fmaxf(m, __shfl_down(m, off, 32));
    if ((tid & 31) == 0) redm[tid >> 5] = m;
    __syncthreads();
    m = redm[0];
#pragma unroll
    for (int i = 1; i < 8; ++i) m = fmaxf(m, redm[i]);

    float s = 0.f;
    for (int i = tid; i < L_; i += 256) s += expf(row[i] - m);
    s = waveReduceAdd(s);
    if ((tid & 31) == 0) reds[tid >> 5] = s;
    __syncthreads();
    s = 0.f;
#pragma unroll
    for (int i = 0; i < 8; ++i) s += reds[i];
    const float inv = 1.f / s;

    float* pr = p + (size_t)b * L_;
    for (int i = tid; i < L_; i += 256) pr[i] = expf(row[i] - m) * inv;
}

// ---------------------------------------------------------------------------
// K3: pointer_weights[b,s,:] = p[b,:]  (bandwidth-bound broadcast, float4)
// ---------------------------------------------------------------------------
__global__ __launch_bounds__(256)
void pgn_broadcast_kernel(const float* __restrict__ p, float* __restrict__ pw) {
    const size_t f = (size_t)blockIdx.x * 256 + threadIdx.x;  // float4 index
    const int l4 = (int)(f & (L_ / 4 - 1));
    const size_t rest = f >> 9;                                // b*S + s
    const int b = (int)(rest >> 10);
    const float4 v = ((const float4*)p)[(size_t)b * (L_ / 4) + l4];
    ((float4*)pw)[f] = v;
}

// ---------------------------------------------------------------------------
// K4: context[b] = P[b] (S x L, f32->bf16) x E[b] (L x D, f32->bf16), f32 acc
// 128x128 workgroup tile, KT=32, 8 waves (2x4), 4x2 WMMA frags per wave.
// Double-buffered LDS, one barrier per K-step. LDS row strides padded
// (A: 40 elems = 20 DW, B: 36 elems = 18 DW) so both staging stores and
// fragment loads hit distinct banks (64 banks x 4B).
// ---------------------------------------------------------------------------
__global__ __launch_bounds__(256)
void pgn_context_wmma_kernel(const float* __restrict__ pw,   // [B,S,L]
                             const float* __restrict__ enc,  // [B,L,D]
                             float* __restrict__ ctx)        // [B,S,D]
{
    constexpr int KT  = 32;
    constexpr int APAD = 40;   // A row stride (elements): 80B rows, 16B-aligned
    constexpr int BPAD = 36;   // B row stride (elements): 72B rows, 8B-aligned
    __shared__ __align__(16) __bf16 Asm[2][128][APAD];  // [buf][m][k]
    __shared__ __align__(16) __bf16 Bsm[2][128][BPAD];  // [buf][n][k] (transposed)

    const int tid = threadIdx.x;
    const int b   = blockIdx.z;
    const int Mb  = blockIdx.y * 128;
    const int Nb  = blockIdx.x * 128;

    const float* Ab = pw  + ((size_t)b * S_) * L_;
    const float* Bb = enc + ((size_t)b * L_) * D_;

    const int lane = tid & 31;
    const int wv   = tid >> 5;     // 0..7
    const int wm   = wv >> 2;      // 0..1  (M wave grid)
    const int wn   = wv & 3;       // 0..3  (N wave grid)
    const int lm   = lane & 15;    // row (A) / col (B) within 16
    const int hh   = lane >> 4;    // K-half select

    // ---- A staging: thread -> 4 float4 slots of the 128x32 tile
    int aRow[4], aCol[4];
    float4 aReg[4];
#pragma unroll
    for (int i = 0; i < 4; ++i) {
        int idx = tid + 256 * i;               // 0..1023 float4 slots
        aRow[i] = idx >> 3;  aCol[i] = (idx & 7) * 4;   // 128 rows x 8 float4
    }
    // ---- B staging: thread -> column n = tid&127, k-block of 16 (kh = tid>>7)
    const int bn  = tid & 127;
    const int bkh = tid >> 7;                  // k base = 16*bkh
    float bRegS[16];

    auto loadA = [&](int k0) {
#pragma unroll
        for (int i = 0; i < 4; ++i)
            aReg[i] = *(const float4*)(Ab + (size_t)(Mb + aRow[i]) * L_ + k0 + aCol[i]);
    };
    auto loadB = [&](int k0) {
#pragma unroll
        for (int e = 0; e < 16; ++e)           // lanes sweep consecutive n: coalesced b32
            bRegS[e] = Bb[(size_t)(k0 + 16 * bkh + e) * D_ + Nb + bn];
    };
    auto storeTiles = [&](int buf) {
#pragma unroll
        for (int i = 0; i < 4; ++i) {          // A: packed pairs -> ds_store_b64
            Asm[buf][aRow[i]][aCol[i] + 0] = f2bf(aReg[i].x);
            Asm[buf][aRow[i]][aCol[i] + 1] = f2bf(aReg[i].y);
            Asm[buf][aRow[i]][aCol[i] + 2] = f2bf(aReg[i].z);
            Asm[buf][aRow[i]][aCol[i] + 3] = f2bf(aReg[i].w);
        }
#pragma unroll
        for (int c = 0; c < 4; ++c) {          // B: k-contiguous bf16x4 -> ds_store_b64
            bf16x4 v4;
#pragma unroll
            for (int e = 0; e < 4; ++e) v4[e] = f2bf(bRegS[4 * c + e]);
            *(bf16x4*)&Bsm[buf][bn][16 * bkh + 4 * c] = v4;
        }
    };

    v8f acc[4][2];
#pragma unroll
    for (int i = 0; i < 4; ++i)
#pragma unroll
        for (int j = 0; j < 2; ++j)
#pragma unroll
            for (int e = 0; e < 8; ++e) acc[i][j][e] = 0.f;

    loadA(0);  loadB(0);
    storeTiles(0);                 // stage K-tile 0 into buffer 0
    loadA(KT); loadB(KT);          // prefetch K-tile 1 into registers

    int buf = 0;
    for (int k0 = 0; k0 < L_; k0 += KT) {
        __syncthreads();    // LDS[buf] fully populated; LDS[buf^1] fully consumed
        if (k0 + KT < L_)       storeTiles(buf ^ 1);                  // stage next tile
        if (k0 + 2 * KT < L_) { loadA(k0 + 2 * KT); loadB(k0 + 2 * KT); }

        // A frag: lane(m=lm, hh): K = 16*hh .. 16*hh+15 (contiguous)
        v16bf afr[4];
#pragma unroll
        for (int i = 0; i < 4; ++i) {
            const int row = wm * 64 + i * 16 + lm;
            bf16x8 c0 = *(const bf16x8*)&Asm[buf][row][16 * hh];
            bf16x8 c1 = *(const bf16x8*)&Asm[buf][row][16 * hh + 8];
#pragma unroll
            for (int e = 0; e < 8; ++e) { afr[i][e] = c0[e]; afr[i][e + 8] = c1[e]; }
        }
        // B frag: lane(n=lm, hh): K = 16*hh .. 16*hh+15 (contiguous; 4x b64, 8B-aligned)
        v16bf bfr[2];
#pragma unroll
        for (int j = 0; j < 2; ++j) {
            const int nn = wn * 32 + j * 16 + lm;
            bf16x4 q0 = *(const bf16x4*)&Bsm[buf][nn][16 * hh + 0];
            bf16x4 q1 = *(const bf16x4*)&Bsm[buf][nn][16 * hh + 4];
            bf16x4 q2 = *(const bf16x4*)&Bsm[buf][nn][16 * hh + 8];
            bf16x4 q3 = *(const bf16x4*)&Bsm[buf][nn][16 * hh + 12];
#pragma unroll
            for (int e = 0; e < 4; ++e) {
                bfr[j][e]      = q0[e];
                bfr[j][e + 4]  = q1[e];
                bfr[j][e + 8]  = q2[e];
                bfr[j][e + 12] = q3[e];
            }
        }
#pragma unroll
        for (int i = 0; i < 4; ++i)
#pragma unroll
            for (int j = 0; j < 2; ++j)
                acc[i][j] = __builtin_amdgcn_wmma_f32_16x16x32_bf16(
                    false, afr[i], false, bfr[j], (short)0, acc[i][j], false, false);
        buf ^= 1;
    }

    // C/D layout: VGPR r -> lanes 0-15: (M=r, N=lane); lanes 16-31: (M=r+8, N=lane-16)
    float* Cb = ctx + ((size_t)b * S_) * D_;
#pragma unroll
    for (int i = 0; i < 4; ++i) {
        const int rbase = Mb + wm * 64 + i * 16 + 8 * hh;
#pragma unroll
        for (int j = 0; j < 2; ++j) {
            const int col = Nb + wn * 32 + j * 16 + lm;
#pragma unroll
            for (int r = 0; r < 8; ++r)
                Cb[(size_t)(rbase + r) * D_ + col] = acc[i][j][r];
        }
    }
}

// ---------------------------------------------------------------------------
// K5: p_gen[b,s] = sigmoid(ctx[b,s,:].w_pgen[D:2D] + dec_pgen[b,s] + b_pgen)
// ---------------------------------------------------------------------------
__global__ __launch_bounds__(128)
void pgn_pgen_kernel(const float* __restrict__ ctx,
                     const float* __restrict__ wpg,
                     const float* __restrict__ dpg,
                     const float* __restrict__ bpg,
                     float* __restrict__ out_pg) {
    const int row = blockIdx.x;            // b*S + s
    const int tid = threadIdx.x;
    const float* r = ctx + (size_t)row * D_;
    const float* w = wpg + D_;
    float acc = 0.f;
#pragma unroll
    for (int j = 0; j < D_ / 128; ++j) {
        int idx = j * 128 + tid;
        acc += r[idx] * w[idx];
    }
    acc = waveReduceAdd(acc);
    __shared__ float red[4];
    if ((tid & 31) == 0) red[tid >> 5] = acc;
    __syncthreads();
    if (tid == 0) {
        float x = red[0] + red[1] + red[2] + red[3] + dpg[row] + bpg[0];
        out_pg[row] = 1.f / (1.f + expf(-x));
    }
}

// ---------------------------------------------------------------------------
extern "C" void kernel_launch(void* const* d_in, const int* in_sizes, int n_in,
                              void* d_out, int out_size, void* d_ws, size_t ws_size,
                              hipStream_t stream) {
    (void)in_sizes; (void)n_in; (void)out_size; (void)ws_size;

    const float* dec  = (const float*)d_in[0];  // [B,S,D]
    const float* enc  = (const float*)d_in[1];  // [B,L,D]
    const int*   mask = (const int*)d_in[2];    // [B,L]
    // d_in[3] = input_ids (unused by the reference outputs)
    const float* wptr = (const float*)d_in[4];  // [2D]
    const float* bptr = (const float*)d_in[5];  // [1]
    const float* wpg  = (const float*)d_in[6];  // [3D]
    const float* bpg  = (const float*)d_in[7];  // [1]

    float* out_pw  = (float*)d_out;                         // [B,S,L]
    float* out_pg  = out_pw + (size_t)B_ * S_ * L_;         // [B,S,1]
    float* out_ctx = out_pg + (size_t)B_ * S_;              // [B,S,D]

    float* ws_es  = (float*)d_ws;            // [B,L]
    float* ws_p   = ws_es + (size_t)B_ * L_; // [B,L]
    float* ws_dpg = ws_p  + (size_t)B_ * L_; // [B,S]

    pgn_dec_pgen_kernel<<<B_ * S_, 128, 0, stream>>>(dec, wpg, ws_dpg);
    pgn_enc_score_kernel<<<B_ * L_, 128, 0, stream>>>(enc, wptr, mask, bptr, ws_es);
    pgn_softmax_kernel<<<B_, 256, 0, stream>>>(ws_es, ws_p);
    pgn_broadcast_kernel<<<(int)(((size_t)B_ * S_ * L_ / 4) / 256), 256, 0, stream>>>(ws_p, out_pw);
    pgn_context_wmma_kernel<<<dim3(D_ / 128, S_ / 128, B_), 256, 0, stream>>>(out_pw, enc, out_ctx);
    pgn_pgen_kernel<<<B_ * S_, 128, 0, stream>>>(out_ctx, wpg, ws_dpg, bpg, out_pg);
}
